// DiffusionFlow_55903294325386
// MI455X (gfx1250) — matrix operand order repaired and checked by reading, not compiled, not run on hardware
//
#include <hip/hip_runtime.h>
#include <hip/hip_bf16.h>

typedef __attribute__((ext_vector_type(16))) _Float16 v16h;
typedef __attribute__((ext_vector_type(8)))  _Float16 v8h;
typedef __attribute__((ext_vector_type(8)))  float    v8f;

#define HID    512
#define NSTEP  10
#define MT     16          // batch rows per workgroup
#define MR     48          // 3*MT rows: [primal | d/dz0 tangent | d/dz1 tangent]
#define SA     520         // padded LDS row stride (halves): 1040B -> good bank spread
#define HSTEP  0.1f

// ---------------- one-time prep kernels (weights -> swizzled f16, c_k) ----------------

// W (512x512 row-major, out x in) -> B = W^T (K x N) in WMMA-B fragment order:
// block (nt*16 + kt) of 32 lanes x 16 halves; lane = n%16 + 16*((k>>3)&1), halves = (k&7) + 8*((k>>4)&1).
__global__ void swizzleW_kernel(const float* __restrict__ W, _Float16* __restrict__ dst) {
  int idx = blockIdx.x * 256 + threadIdx.x;     // 512*512
  if (idx >= HID * HID) return;
  int n = idx >> 9, k = idx & 511;
  int off = (((n >> 4) * 16 + (k >> 5)) * 32 + (n & 15) + (((k >> 3) & 1) << 4)) * 16 + (k & 7) + (((k >> 4) & 1) << 3);
  dst[off] = (_Float16)W[n * HID + k];
}

// W3 (2x512) -> padded 512x16 B fragments (cols >=2 are zero)
__global__ void swizzleW3_kernel(const float* __restrict__ W3, _Float16* __restrict__ dst) {
  int idx = blockIdx.x * 256 + threadIdx.x;     // 512*16
  if (idx >= HID * 16) return;
  int k = idx >> 4, j = idx & 15;
  float v = (j < 2) ? W3[j * HID + k] : 0.f;
  dst[((k >> 5) * 32 + j + (((k >> 3) & 1) << 4)) * 16 + (k & 7) + (((k >> 4) & 1) << 3)] = (_Float16)v;
}

// c_k[n] = b0[n] + sum_j W0[n][2+j] * temb_k[j]  (t uniform per step!)
__global__ void cks_kernel(const float* __restrict__ W0, const float* __restrict__ b0,
                           float* __restrict__ cks) {
  int idx = blockIdx.x * 256 + threadIdx.x;     // 10*512
  if (idx >= NSTEP * HID) return;
  int ks = idx >> 9, n = idx & 511;
  float t = ks * HSTEP;
  float acc = b0[n];
  for (int j = 0; j < 16; ++j) {
    float f = __expf(-__logf(10000.f) * (float)j / 16.f);
    float a = t * f;
    acc += W0[n * 34 + 2 + j]      * __sinf(a);   // sin half
    acc += W0[n * 34 + 2 + 16 + j] * __cosf(a);   // cos half
  }
  cks[idx] = acc;
}

// ---------------- main fused flow kernel ----------------

// A fragment per ISA 16-bit A layout: lane<16 -> halves = K[kb..kb+7],K[kb+16..kb+23]
__device__ __forceinline__ v16h ldsFragA(const _Float16* A, int row, int kbase) {
  v8h lo = *(const v8h*)(A + row * SA + kbase);
  v8h hi = *(const v8h*)(A + row * SA + kbase + 16);
  return __builtin_shufflevector(lo, hi, 0,1,2,3,4,5,6,7,8,9,10,11,12,13,14,15);
}

#define WMMA16(A, B, C) \
  __builtin_amdgcn_wmma_f32_16x16x32_f16(false, (A), false, (B), (short)0, (C), false, false)

// One 512->512 layer for the stacked [primal;tan0;tan1] 48x512 matrix.
// N-register-blocked: each wave holds 4 N-tiles x 3 passes = 12 accumulator tiles,
// loads the 3 shared A fragments ONCE per K-step, streams 4 B fragments against them.
__device__ __forceinline__ void layer512(const _Float16* __restrict__ As,
                                         _Float16* __restrict__ Ad,
                                         const _Float16* __restrict__ Bsw,
                                         const float* __restrict__ bias,
                                         int wave, int lane)
{
  const int rlo  = lane & 15;           // row-in-tile for A / col-in-tile for C
  const int kgrp = (lane >> 4) * 8;     // A K sub-offset per lane half
  const int mofs = (lane >> 4) * 8;     // C/D row offset per lane half

  v8f accP[4]  = {};
  v8f accT0[4] = {};
  v8f accT1[4] = {};
  const _Float16* bp = Bsw + (wave * 4) * 16 * 512 + lane * 16;  // contiguous 32B/lane

#pragma unroll 2
  for (int kt = 0; kt < 16; ++kt) {
    const int kb = kt * 32 + kgrp;
    v16h aP  = ldsFragA(As, rlo,      kb);
    v16h aT0 = ldsFragA(As, rlo + 16, kb);
    v16h aT1 = ldsFragA(As, rlo + 32, kb);
#pragma unroll
    for (int g = 0; g < 4; ++g) {
      v16h b = *(const v16h*)(bp + (g * 16 + kt) * 512);
      accP[g]  = WMMA16(aP,  b, accP[g]);
      accT0[g] = WMMA16(aT0, b, accT0[g]);
      accT1[g] = WMMA16(aT1, b, accT1[g]);
    }
  }

#pragma unroll
  for (int g = 0; g < 4; ++g) {
    const int n  = (wave * 4 + g) * 16 + rlo;
    const float bn = bias[n];
#pragma unroll
    for (int r = 0; r < 8; ++r) {
      const int row = mofs + r;                     // batch row in tile
      float u  = accP[g][r] + bn;
      float s  = 1.f / (1.f + __expf(-u));
      float a  = u * s;                             // silu(u)
      float dg = s * (1.f + u * (1.f - s));         // silu'(u)
      Ad[row * SA + n]        = (_Float16)a;
      Ad[(16 + row) * SA + n] = (_Float16)(accT0[g][r] * dg);
      Ad[(32 + row) * SA + n] = (_Float16)(accT1[g][r] * dg);
    }
  }
}

__global__ void __launch_bounds__(256, 1)
flow_kernel(const float* __restrict__ x,
            const float* __restrict__ W0,
            const _Float16* __restrict__ B1sw, const float* __restrict__ b1,
            const _Float16* __restrict__ B2sw, const float* __restrict__ b2,
            const _Float16* __restrict__ B3sw, const float* __restrict__ b3,
            const float* __restrict__ cks,
            float* __restrict__ out)
{
  __shared__ __align__(16) _Float16 sA[MR * SA];
  __shared__ __align__(16) _Float16 sB[MR * SA];
  __shared__ float sV[MR * 2];
  __shared__ float sZ[MT * 2];
  __shared__ float sLd[MT];

  const int tid   = threadIdx.x;
  const int wave  = tid >> 5;
  const int lane  = tid & 31;
  const int gbase = blockIdx.x * MT;

  if (tid < MT) {
    sZ[tid * 2 + 0] = x[(gbase + tid) * 2 + 0];
    sZ[tid * 2 + 1] = x[(gbase + tid) * 2 + 1];
    sLd[tid] = 0.f;
  }
  __syncthreads();

  for (int k = 0; k < NSTEP; ++k) {
    const float* ck = cks + k * HID;
    // ---- layer 0 (K=2 + per-step constant): pure VALU, writes a0 & gated tangents
    for (int idx = tid; idx < MT * HID; idx += 256) {
      int m = idx >> 9, n = idx & 511;
      float w0 = W0[n * 34 + 0], w1 = W0[n * 34 + 1];
      float u  = sZ[m * 2] * w0 + sZ[m * 2 + 1] * w1 + ck[n];
      float s  = 1.f / (1.f + __expf(-u));
      float dg = s * (1.f + u * (1.f - s));
      sA[m * SA + n]        = (_Float16)(u * s);
      sA[(16 + m) * SA + n] = (_Float16)(dg * w0);   // tangent e0: du0 = W0[:,0]
      sA[(32 + m) * SA + n] = (_Float16)(dg * w1);   // tangent e1: du0 = W0[:,1]
    }
    __syncthreads();
    layer512(sA, sB, B1sw, b1, wave, lane);
    __syncthreads();
    layer512(sB, sA, B2sw, b2, wave, lane);
    __syncthreads();
    // ---- layer 3: 512 -> 2 (N padded to 16); wave w handles pass w (primal/tan0/tan1)
    if (wave < 3) {
      const int rlo  = lane & 15;
      const int kgrp = (lane >> 4) * 8;
      const int mofs = (lane >> 4) * 8;
      v8f acc = {};
      const _Float16* bp = B3sw + lane * 16;
#pragma unroll 4
      for (int kt = 0; kt < 16; ++kt) {
        v16h b = *(const v16h*)(bp + kt * 512);
        v16h a = ldsFragA(sA, wave * 16 + rlo, kt * 32 + kgrp);
        acc = WMMA16(a, b, acc);
      }
      if (rlo < 2) {
        float bv = b3[rlo];
#pragma unroll
        for (int r = 0; r < 8; ++r)
          sV[(wave * 16 + mofs + r) * 2 + rlo] = acc[r] + bv;
      }
    }
    __syncthreads();
    // ---- det / log-det / Euler update (2x2 Jacobian per point)
    if (tid < MT) {
      float v0  = sV[tid * 2], v1 = sV[tid * 2 + 1];
      float a00 = sV[(16 + tid) * 2 + 0];   // dv0/dz0
      float a10 = sV[(16 + tid) * 2 + 1];   // dv1/dz0
      float a01 = sV[(32 + tid) * 2 + 0];   // dv0/dz1
      float a11 = sV[(32 + tid) * 2 + 1];   // dv1/dz1
      float det = (1.f + HSTEP * a00) * (1.f + HSTEP * a11) - HSTEP * HSTEP * a01 * a10;
      sLd[tid] += __logf(fmaxf(fabsf(det), 1e-8f));
      sZ[tid * 2 + 0] += HSTEP * v0;
      sZ[tid * 2 + 1] += HSTEP * v1;
    }
    __syncthreads();
  }

  if (tid < MT) {
    float z0 = sZ[tid * 2], z1 = sZ[tid * 2 + 1];
    float logpz = -0.5f * (z0 * z0 + z1 * z1) - __logf(2.f * 3.14159265358979f);
    out[gbase + tid] = logpz + sLd[tid];
  }
}

// ---------------- launch ----------------

extern "C" void kernel_launch(void* const* d_in, const int* in_sizes, int n_in,
                              void* d_out, int out_size, void* d_ws, size_t ws_size,
                              hipStream_t stream) {
  (void)in_sizes; (void)n_in; (void)ws_size;
  const float* x  = (const float*)d_in[0];
  const float* W0 = (const float*)d_in[1];
  const float* b0 = (const float*)d_in[2];
  const float* W1 = (const float*)d_in[3];
  const float* b1 = (const float*)d_in[4];
  const float* W2 = (const float*)d_in[5];
  const float* b2 = (const float*)d_in[6];
  const float* W3 = (const float*)d_in[7];
  const float* b3 = (const float*)d_in[8];

  // workspace layout: swizzled f16 weights + per-step layer0 constants
  _Float16* B1sw = (_Float16*)d_ws;
  _Float16* B2sw = B1sw + HID * HID;
  _Float16* B3sw = B2sw + HID * HID;
  float*    cks  = (float*)(B3sw + HID * 16);   // byte offset 1,064,960 (4B aligned)

  swizzleW_kernel <<<(HID * HID + 255) / 256, 256, 0, stream>>>(W1, B1sw);
  swizzleW_kernel <<<(HID * HID + 255) / 256, 256, 0, stream>>>(W2, B2sw);
  swizzleW3_kernel<<<(HID * 16  + 255) / 256, 256, 0, stream>>>(W3, B3sw);
  cks_kernel      <<<(NSTEP * HID + 255) / 256, 256, 0, stream>>>(W0, b0, cks);

  const int nblocks = out_size / MT;   // 131072 / 16 = 8192 workgroups
  flow_kernel<<<nblocks, 256, 0, stream>>>(x, W0, B1sw, b1, B2sw, b2, B3sw, b3, cks,
                                           (float*)d_out);
}
